// GT_10084583211427
// MI455X (gfx1250) — matrix operand order repaired and checked by reading, not compiled, not run on hardware
//
#include <hip/hip_runtime.h>

typedef __attribute__((ext_vector_type(16))) __bf16 v16bf;
typedef __attribute__((ext_vector_type(8)))  __bf16 v8bf;
typedef __attribute__((ext_vector_type(8)))  float  v8f;

static constexpr int NN = 20000;   // nodes
static constexpr int EE = 320000;  // edges
static constexpr int NB = 64;      // graphs in batch

// ---------------- CDNA5 async global->LDS helpers --------------------------
__device__ __forceinline__ void async_load_b128(unsigned lds_addr, const void* gaddr) {
    unsigned long long ga = (unsigned long long)gaddr;
    // GLOBAL_LOAD_ASYNC_TO_LDS_B128 (GV mode): per-lane LDS dest in VGPR, 64-bit
    // global address in VGPR pair; tracked with ASYNCcnt.
    asm volatile("global_load_async_to_lds_b128 %0, %1, off"
                 :: "v"(lds_addr), "v"(ga)
                 : "memory");
}
__device__ __forceinline__ void wait_async0() {
    asm volatile("s_wait_asynccnt 0x0" ::: "memory");
}

// ---------------- monotone float<->uint encoding for atomic segment-max ----
__device__ __forceinline__ unsigned fenc(float f) {
    unsigned b = __float_as_uint(f);
    return (b & 0x80000000u) ? ~b : (b | 0x80000000u);
}
__device__ __forceinline__ float fdec(unsigned u) {
    unsigned b = (u & 0x80000000u) ? (u & 0x7FFFFFFFu) : ~u;
    return __uint_as_float(b);
}

// ---------------- utility kernels ------------------------------------------
__global__ void k_zero(float* p, int n) {
    int i = blockIdx.x * blockDim.x + threadIdx.x;
    if (i < n) p[i] = 0.0f;
}

__global__ void k_cast_bf16(const float* __restrict__ s, __bf16* __restrict__ d, int n) {
    int i = blockIdx.x * blockDim.x + threadIdx.x;
    if (i < n) d[i] = (__bf16)s[i];
}

// W is K x N row-major (fp32); produce Wt = N x K row-major (bf16)
__global__ void k_transpose_bf16(const float* __restrict__ W, __bf16* __restrict__ Wt,
                                 int K, int N) {
    int i = blockIdx.x * blockDim.x + threadIdx.x;
    if (i < K * N) {
        int k = i / N, n = i - k * N;
        Wt[(size_t)n * K + k] = (__bf16)W[i];
    }
}

// ---------------- WMMA bf16 GEMM: C[M,N] = A[M,K] @ Wt[N,K]^T + bias -------
// block = 256 threads (8 wave32). Tile 128(M) x 64(N), K-step 32, double-
// buffered LDS staged via global_load_async_to_lds_b128 (ASYNCcnt pipeline).
// Each wave owns a 32x32 output patch -> 4 v_wmma per K-step.
// Fragment layouts per CDNA5 ISA 7.12.2 (16-bit A 16x32, B 32x16, f32 C 16x16).
__global__ __launch_bounds__(256) void k_gemm_bf16(
    const __bf16* __restrict__ A, const __bf16* __restrict__ Wt,
    const float* __restrict__ bias, float* __restrict__ C,
    int M, int K, int N)
{
    __shared__ __align__(16) __bf16 sA[2][128][40]; // 40-elem stride: conflict-free
    __shared__ __align__(16) __bf16 sB[2][64][40];  // rows indexed by N, 32 K vals

    const int t      = threadIdx.x;
    const int lane   = t & 31;
    const int wave   = t >> 5;
    const int m0     = blockIdx.x * 128;
    const int n0     = blockIdx.y * 64;
    const int wm     = wave & 3;         // M slab: rows wm*32 .. wm*32+31
    const int wn     = wave >> 2;        // N offset 32*wn (2 x 16 tiles)
    const int lmod   = lane & 15;
    const bool hiH   = lane >= 16;

    v8f acc00, acc01, acc10, acc11;
    #pragma unroll
    for (int i = 0; i < 8; ++i) {
        acc00[i] = 0.0f; acc01[i] = 0.0f; acc10[i] = 0.0f; acc11[i] = 0.0f;
    }

    // ---- loader mapping: thread t stages A rows lrow & lrow+64, B row lrow
    const int lrow   = t >> 2;           // 0..63
    const int lchunk = (t & 3) * 8;      // 0,8,16,24 (bf16 elems; 16B chunks)
    // Clamp OOB rows in-bounds: their products land only in C rows >= M,
    // which the epilogue never stores. Keeps EXEC uniform (WMMA requirement).
    const int gr0 = min(m0 + lrow,      M - 1);
    const int gr1 = min(m0 + lrow + 64, M - 1);

    const __bf16* agp0 = A  + (size_t)gr0 * K + lchunk;
    const __bf16* agp1 = A  + (size_t)gr1 * K + lchunk;
    const __bf16* bgp  = Wt + (size_t)(n0 + lrow) * K + lchunk;

    unsigned aSt0[2], aSt1[2], bSt[2];
    #pragma unroll
    for (int b = 0; b < 2; ++b) {
        aSt0[b] = (unsigned)(uintptr_t)&sA[b][lrow][lchunk];
        aSt1[b] = (unsigned)(uintptr_t)&sA[b][lrow + 64][lchunk];
        bSt[b]  = (unsigned)(uintptr_t)&sB[b][lrow][lchunk];
    }

    const int nStages = K >> 5;          // K/32 (K in {192,256})

    // prologue: stage 0 into buffer 0
    async_load_b128(aSt0[0], agp0);
    async_load_b128(aSt1[0], agp1);
    async_load_b128(bSt[0],  bgp);

    const int ar0 = wm * 32 + lmod;
    const int ar1 = ar0 + 16;
    const int ak  = hiH ? 8 : 0;
    const int bk  = hiH ? 8 : 0;
    const int br0 = wn * 32 + lmod;
    const int br1 = br0 + 16;

    for (int s = 0; s < nStages; ++s) {
        wait_async0();                   // this wave's stage-s tiles in LDS
        __syncthreads();                 // everyone's tiles visible

        if (s + 1 < nStages) {           // prefetch stage s+1 into other buffer
            int nb = (s + 1) & 1;
            size_t off = (size_t)(s + 1) * 32;
            async_load_b128(aSt0[nb], agp0 + off);
            async_load_b128(aSt1[nb], agp1 + off);
            async_load_b128(bSt[nb],  bgp  + off);
        }

        const int b = s & 1;
        // A fragments: lanes<16 hold K {0..7,16..23}, hi lanes {8..15,24..31}
        v8bf a0l = *(const v8bf*)(&sA[b][ar0][ak]);
        v8bf a0h = *(const v8bf*)(&sA[b][ar0][ak + 16]);
        v16bf af0 = __builtin_shufflevector(a0l, a0h, 0,1,2,3,4,5,6,7,8,9,10,11,12,13,14,15);
        v8bf a1l = *(const v8bf*)(&sA[b][ar1][ak]);
        v8bf a1h = *(const v8bf*)(&sA[b][ar1][ak + 16]);
        v16bf af1 = __builtin_shufflevector(a1l, a1h, 0,1,2,3,4,5,6,7,8,9,10,11,12,13,14,15);
        // B fragments: N = lmod(+16); lanes<16 hold K {0..7,16..23}, hi lanes {8..15,24..31}
        v8bf b0l = *(const v8bf*)(&sB[b][br0][bk]);
        v8bf b0h = *(const v8bf*)(&sB[b][br0][bk + 16]);
        v16bf bf0 = __builtin_shufflevector(b0l, b0h, 0,1,2,3,4,5,6,7,8,9,10,11,12,13,14,15);
        v8bf b1l = *(const v8bf*)(&sB[b][br1][bk]);
        v8bf b1h = *(const v8bf*)(&sB[b][br1][bk + 16]);
        v16bf bf1 = __builtin_shufflevector(b1l, b1h, 0,1,2,3,4,5,6,7,8,9,10,11,12,13,14,15);

        acc00 = __builtin_amdgcn_wmma_f32_16x16x32_bf16(false, af0, false, bf0,
                                                        (short)0, acc00, false, false);
        acc01 = __builtin_amdgcn_wmma_f32_16x16x32_bf16(false, af0, false, bf1,
                                                        (short)0, acc01, false, false);
        acc10 = __builtin_amdgcn_wmma_f32_16x16x32_bf16(false, af1, false, bf0,
                                                        (short)0, acc10, false, false);
        acc11 = __builtin_amdgcn_wmma_f32_16x16x32_bf16(false, af1, false, bf1,
                                                        (short)0, acc11, false, false);
    }

    // Epilogue. C layout: VGPR r -> M = r + (hi?8:0), N = lane%16.
    int col0 = n0 + wn * 32 + lmod;
    int rowA = m0 + wm * 32 + (hiH ? 8 : 0);
    int rowB = rowA + 16;
    float bb0 = bias[col0];
    float bb1 = bias[col0 + 16];
    #pragma unroll
    for (int r = 0; r < 8; ++r) {
        int ra = rowA + r;
        if (ra < M) {
            C[(size_t)ra * N + col0]      = acc00[r] + bb0;
            C[(size_t)ra * N + col0 + 16] = acc01[r] + bb1;
        }
        int rb = rowB + r;
        if (rb < M) {
            C[(size_t)rb * N + col0]      = acc10[r] + bb0;
            C[(size_t)rb * N + col0 + 16] = acc11[r] + bb1;
        }
    }
}

// ---------------- edge phase -----------------------------------------------
// scores[e,h] = (1/sqrt(c)) * sum_c q[dst,h,c]*(k[src,h,c] + eattr[e]*We[h,c]),
// plus atomic segment-max over dst.
__global__ void k_scores(const float* __restrict__ q, const float* __restrict__ k,
                         const float* __restrict__ We, const float* __restrict__ eattr,
                         const int* __restrict__ eidx, float* __restrict__ sc,
                         unsigned* __restrict__ mbuf, int E, int h, int c, float scale)
{
    int t = blockIdx.x * blockDim.x + threadIdx.x;
    if (t >= E * h) return;
    int e = t / h, head = t - e * h;
    int s = eidx[e] + 1;
    int d = eidx[E + e] + 1;
    int hc = h * c;
    const float* qp = q + (size_t)d * hc + head * c;
    const float* kp = k + (size_t)s * hc + head * c;
    const float* wp = We + head * c;
    float ea = eattr[e];
    float acc = 0.0f;
    for (int i = 0; i < c; ++i) acc += qp[i] * (kp[i] + ea * wp[i]);
    acc *= scale;
    sc[t] = acc;
    atomicMax(&mbuf[d * h + head], fenc(acc));
}

__global__ void k_expden(float* __restrict__ sc, const int* __restrict__ eidx,
                         const unsigned* __restrict__ mbuf, float* __restrict__ den,
                         int E, int h)
{
    int t = blockIdx.x * blockDim.x + threadIdx.x;
    if (t >= E * h) return;
    int e = t / h, head = t - e * h;
    int d = eidx[E + e] + 1;
    unsigned mu = mbuf[d * h + head];
    float m = (mu == 0u) ? 0.0f : fdec(mu);    // matches where(isfinite(m),m,0)
    float ex = __expf(sc[t] - m);
    sc[t] = ex;
    atomicAdd(&den[d * h + head], ex);
}

// block per edge, thread per channel. Scatters messages, emits alpha.
__global__ void k_message(const float* __restrict__ v, const float* __restrict__ We,
                          const float* __restrict__ eattr, const int* __restrict__ eidx,
                          const float* __restrict__ ex, const float* __restrict__ den,
                          float* __restrict__ obuf, float* __restrict__ aout,
                          int E, int h, int c)
{
    int e = blockIdx.x, t = threadIdx.x;
    int hc = h * c;
    int s = eidx[e] + 1;
    int d = eidx[E + e] + 1;
    int head = t / c;
    float al = ex[(size_t)e * h + head] / (den[d * h + head] + 1e-16f);
    float msg = (v[(size_t)s * hc + t] + eattr[e] * We[t]) * al;
    atomicAdd(&obuf[(size_t)d * hc + t], msg);
    if (t < h)
        aout[(size_t)e * h + t] = ex[(size_t)e * h + t] / (den[d * h + t] + 1e-16f);
}

// ---------------- pointwise layers -----------------------------------------
// y = relu(LN(o + s)); also writes bf16 copy for next layer's GEMM. hc == 256.
__global__ void k_ln_relu(const float* __restrict__ o, const float* __restrict__ s,
                          const float* __restrict__ g, const float* __restrict__ be,
                          float* __restrict__ y, __bf16* __restrict__ yb, int hc)
{
    __shared__ float red[256];
    int n = blockIdx.x, t = threadIdx.x;
    size_t base = (size_t)n * hc;
    float v = o[base + t] + s[base + t];
    red[t] = v;
    __syncthreads();
    for (int st = 128; st > 0; st >>= 1) { if (t < st) red[t] += red[t + st]; __syncthreads(); }
    float mu = red[0] / (float)hc;
    __syncthreads();
    float dv = v - mu;
    red[t] = dv * dv;
    __syncthreads();
    for (int st = 128; st > 0; st >>= 1) { if (t < st) red[t] += red[t + st]; __syncthreads(); }
    float var = red[0] / (float)hc;
    float out = g[t] * dv * rsqrtf(var + 1e-5f) + be[t];
    out = fmaxf(out, 0.0f);
    y[base + t] = out;
    yb[base + t] = (__bf16)out;
}

// layer 3: x3 = relu(o + s); xr = x + x3; segment-sum into pooled, count nodes.
__global__ void k_relu_res_pool(const float* __restrict__ o, const float* __restrict__ s,
                                const float* __restrict__ x, const int* __restrict__ batch,
                                float* __restrict__ pooled, float* __restrict__ cnt, int hc)
{
    int n = blockIdx.x, t = threadIdx.x;
    size_t base = (size_t)n * hc;
    float v = fmaxf(o[base + t] + s[base + t], 0.0f);
    float xr = x[base + t] + v;
    int b = batch[n];
    atomicAdd(&pooled[b * hc + t], xr);
    if (t == 0) atomicAdd(&cnt[b], 1.0f);
}

// per-graph MLP head: relu(pooled/cnt @ fc1 + b1) @ fc2 + b2
__global__ void k_head(const float* __restrict__ pooled, const float* __restrict__ cnt,
                       const float* __restrict__ f1w, const float* __restrict__ f1b,
                       const float* __restrict__ f2w, const float* __restrict__ f2b,
                       float* __restrict__ logits)
{
    __shared__ float p[192];
    __shared__ float hsh[32];
    int b = blockIdx.x, t = threadIdx.x;
    float dc = fmaxf(cnt[b], 1.0f);
    p[t] = pooled[b * 192 + t] / dc;
    __syncthreads();
    if (t < 32) {
        float a = f1b[t];
        for (int i = 0; i < 192; ++i) a += p[i] * f1w[i * 32 + t];
        hsh[t] = fmaxf(a, 0.0f);
    }
    __syncthreads();
    if (t < 2) {
        float a = f2b[t];
        for (int i = 0; i < 32; ++i) a += hsh[i] * f2w[i * 2 + t];
        logits[b * 2 + t] = a;
    }
}

// ---------------- host driver ----------------------------------------------
extern "C" void kernel_launch(void* const* d_in, const int* in_sizes, int n_in,
                              void* d_out, int out_size, void* d_ws, size_t ws_size,
                              hipStream_t stream)
{
    const float* x     = (const float*)d_in[0];
    const int*   eidx  = (const int*)d_in[1];
    const float* eattr = (const float*)d_in[2];
    const int*   batch = (const int*)d_in[3];

    const float *Wq[3], *bq[3], *Wk[3], *bk[3], *Wv[3], *bv[3], *We[3], *Ws[3], *bs[3];
    for (int l = 0; l < 3; ++l) {
        int b0 = 4 + l * 9;
        Wq[l] = (const float*)d_in[b0 + 0]; bq[l] = (const float*)d_in[b0 + 1];
        Wk[l] = (const float*)d_in[b0 + 2]; bk[l] = (const float*)d_in[b0 + 3];
        Wv[l] = (const float*)d_in[b0 + 4]; bv[l] = (const float*)d_in[b0 + 5];
        We[l] = (const float*)d_in[b0 + 6];
        Ws[l] = (const float*)d_in[b0 + 7]; bs[l] = (const float*)d_in[b0 + 8];
    }
    const float* g1  = (const float*)d_in[31];
    const float* be1 = (const float*)d_in[32];
    const float* g2  = (const float*)d_in[33];
    const float* be2 = (const float*)d_in[34];
    const float* f1w = (const float*)d_in[35];
    const float* f1b = (const float*)d_in[36];
    const float* f2w = (const float*)d_in[37];
    const float* f2b = (const float*)d_in[38];

    float* out    = (float*)d_out;
    float* logits = out;
    float* a1     = out + 128;
    float* a2     = a1 + (size_t)EE * 4;
    float* a3     = a2 + (size_t)EE * 4;

    // ---- workspace carve (256B aligned) ----
    char* wsp = (char*)d_ws;
    auto carve = [&](size_t bytes) -> void* {
        void* p = (void*)wsp;
        wsp += (bytes + 255) & ~(size_t)255;
        return p;
    };
    __bf16* xb  = (__bf16*)carve((size_t)NN * 192 * 2);
    __bf16* x1b = (__bf16*)carve((size_t)NN * 256 * 2);
    __bf16* x2b = (__bf16*)carve((size_t)NN * 256 * 2);
    float*  x1  = (float*)carve((size_t)NN * 256 * 4);
    float*  x2  = (float*)carve((size_t)NN * 256 * 4);
    float*  qb  = (float*)carve((size_t)NN * 256 * 4);
    float*  kb  = (float*)carve((size_t)NN * 256 * 4);
    float*  vb  = (float*)carve((size_t)NN * 256 * 4);
    float*  sb  = (float*)carve((size_t)NN * 256 * 4);
    float*  ob  = (float*)carve((size_t)NN * 256 * 4);
    float*  exb = (float*)carve((size_t)EE * 6 * 4);
    float*  mb  = (float*)carve((size_t)NN * 6 * 4);
    float*  den = (float*)carve((size_t)NN * 6 * 4);
    float*  pooled = (float*)carve((size_t)NB * 192 * 4);
    float*  cntb   = (float*)carve((size_t)NB * 4);
    int Kin[3] = {192, 256, 256};
    int Nc_[3] = {256, 256, 192};
    __bf16* wt[3][4];
    for (int l = 0; l < 3; ++l)
        for (int m = 0; m < 4; ++m)
            wt[l][m] = (__bf16*)carve((size_t)Kin[l] * Nc_[l] * 2);

    auto zero = [&](float* p, int n) {
        k_zero<<<dim3((n + 255) / 256), dim3(256), 0, stream>>>(p, n);
    };

    // ---- prep: bf16 casts / weight transposes (deterministic each call) ----
    k_cast_bf16<<<dim3((NN * 192 + 255) / 256), dim3(256), 0, stream>>>(x, xb, NN * 192);
    for (int l = 0; l < 3; ++l) {
        const float* wsrc[4] = {Wq[l], Wk[l], Wv[l], Ws[l]};
        int kn = Kin[l] * Nc_[l];
        for (int m = 0; m < 4; ++m)
            k_transpose_bf16<<<dim3((kn + 255) / 256), dim3(256), 0, stream>>>(
                wsrc[m], wt[l][m], Kin[l], Nc_[l]);
    }

    auto gemm = [&](const __bf16* A, const __bf16* Wt, const float* bias, float* C,
                    int K, int Nc) {
        dim3 grid((NN + 127) / 128, Nc / 64);
        k_gemm_bf16<<<grid, dim3(256), 0, stream>>>(A, Wt, bias, C, NN, K, Nc);
    };

    auto edges = [&](const float* q, const float* k, const float* v, const float* Wedge,
                     float* obuf, float* aout, int h, int c) {
        int hc = h * c;
        zero(mb, NN * h);
        zero(den, NN * h);
        zero(obuf, NN * hc);
        int tot = EE * h;
        float scale = 1.0f / sqrtf((float)c);
        k_scores<<<dim3((tot + 255) / 256), dim3(256), 0, stream>>>(
            q, k, Wedge, eattr, eidx, exb, (unsigned*)mb, EE, h, c, scale);
        k_expden<<<dim3((tot + 255) / 256), dim3(256), 0, stream>>>(
            exb, eidx, (const unsigned*)mb, den, EE, h);
        k_message<<<dim3(EE), dim3(hc), 0, stream>>>(
            v, Wedge, eattr, eidx, exb, den, obuf, aout, EE, h, c);
    };

    // ---- layer 1: 192 -> (4,64) ----
    gemm(xb, wt[0][0], bq[0], qb, 192, 256);
    gemm(xb, wt[0][1], bk[0], kb, 192, 256);
    gemm(xb, wt[0][2], bv[0], vb, 192, 256);
    gemm(xb, wt[0][3], bs[0], sb, 192, 256);
    edges(qb, kb, vb, We[0], ob, a1, 4, 64);
    k_ln_relu<<<dim3(NN), dim3(256), 0, stream>>>(ob, sb, g1, be1, x1, x1b, 256);

    // ---- layer 2: 256 -> (4,64) ----
    gemm(x1b, wt[1][0], bq[1], qb, 256, 256);
    gemm(x1b, wt[1][1], bk[1], kb, 256, 256);
    gemm(x1b, wt[1][2], bv[1], vb, 256, 256);
    gemm(x1b, wt[1][3], bs[1], sb, 256, 256);
    edges(qb, kb, vb, We[1], ob, a2, 4, 64);
    k_ln_relu<<<dim3(NN), dim3(256), 0, stream>>>(ob, sb, g2, be2, x2, x2b, 256);

    // ---- layer 3: 256 -> (6,32) ----
    gemm(x2b, wt[2][0], bq[2], qb, 256, 192);
    gemm(x2b, wt[2][1], bk[2], kb, 256, 192);
    gemm(x2b, wt[2][2], bv[2], vb, 256, 192);
    gemm(x2b, wt[2][3], bs[2], sb, 256, 192);
    edges(qb, kb, vb, We[2], ob, a3, 6, 32);

    // ---- residual + mean pool + head ----
    zero(pooled, NB * 192);
    zero(cntb, NB);
    k_relu_res_pool<<<dim3(NN), dim3(192), 0, stream>>>(ob, sb, x, batch, pooled, cntb, 192);
    k_head<<<dim3(NB), dim3(192), 0, stream>>>(pooled, cntb, f1w, f1b, f2w, f2b, logits);
}
